// Attention_47321949667809
// MI455X (gfx1250) — matrix operand order repaired and checked by reading, not compiled, not run on hardware
//
#include <hip/hip_runtime.h>
#include <hip/hip_bf16.h>

// CDNA5 WMMA types
typedef __attribute__((ext_vector_type(16))) __bf16 v16bf;
typedef __attribute__((ext_vector_type(8)))  float  v8f;

union Frag16 { v16bf v; uint4 q[2]; };

static __device__ __forceinline__ v8f v8zero() {
  v8f z = {0.f,0.f,0.f,0.f,0.f,0.f,0.f,0.f};
  return z;
}

#define WMMA_BF16(a, b, c) \
  __builtin_amdgcn_wmma_f32_16x16x32_bf16(false, (a), false, (b), (short)0, (c), false, false)

// Guarded CDNA5 async global->LDS path (falls back to sync copies if the
// toolchain doesn't declare the builtins).
#define AS1 __attribute__((address_space(1)))
#define AS3 __attribute__((address_space(3)))
typedef int v4i_vs __attribute__((vector_size(16)));   // matches builtin param type
#if defined(__has_builtin)
#if __has_builtin(__builtin_amdgcn_global_load_async_to_lds_b128) && \
    __has_builtin(__builtin_amdgcn_s_wait_asynccnt)
#define USE_ASYNC_LDS 1
#endif
#endif
#ifndef USE_ASYNC_LDS
#define USE_ASYNC_LDS 0
#endif

// Problem dims (fixed by reference)
constexpr int Bc = 2, Dd = 2048, Nn = 1024, Hh = 16, Pp = 64;
constexpr int N3 = 3 * Nn;            // 3072
constexpr int LDA = 48;               // LDS row stride (bf16): 96 B, 16B-aligned
constexpr int LDK = 72;               // LDS row stride (bf16): 144 B, 16B-aligned

// --------------------------------------------------------------------------
// Kernel 1: QKV projection.  C(4096 x 3072) = x(4096 x 1024) @ W + bias.
// Workgroup tile 128x128, BK=32, 8 waves (2x4), wave tile 64x32.
// Global->register->LDS double buffering: next tile's global loads are in
// flight while the current tile's 8 WMMAs execute.
// --------------------------------------------------------------------------
__global__ __launch_bounds__(256) void qkv_kernel(
    const float* __restrict__ x, const float* __restrict__ W,
    const float* __restrict__ bias,
    __bf16* __restrict__ Qb, __bf16* __restrict__ Kb, __bf16* __restrict__ Vb)
{
  __shared__ __align__(16) __bf16 sA[128][LDA];   // [m][k]
  __shared__ __align__(16) __bf16 sB[128][LDA];   // transposed: [n][k]

  const int tid  = threadIdx.x;
  const int lane = tid & 31;
  const int wid  = tid >> 5;
  const int wm   = (wid >> 2) * 64;     // 0 or 64
  const int wn   = (wid & 3) * 32;      // 0,32,64,96
  const int bm   = blockIdx.y * 128;
  const int bn   = blockIdx.x * 128;

  v8f acc[4][2];
  #pragma unroll
  for (int i = 0; i < 4; ++i)
    #pragma unroll
    for (int j = 0; j < 2; ++j) acc[i][j] = v8zero();

  // register staging buffers (double buffer through VGPRs)
  float4 ra[4], rb[4];
  #pragma unroll
  for (int i = 0; i < 4; ++i) {
    int q = tid + 256 * i;
    ra[i] = *(const float4*)(x + (size_t)(bm + (q >> 3)) * Nn + (q & 7) * 4);
    rb[i] = *(const float4*)(W + (size_t)(q >> 5) * N3 + bn + (q & 31) * 4);
  }

  for (int kt = 0; kt < Nn / 32; ++kt) {
    __syncthreads();   // previous tile's LDS readers are done
    // Publish staged tile (fp32 -> bf16)
    #pragma unroll
    for (int i = 0; i < 4; ++i) {
      int q   = tid + 256 * i;
      int row = q >> 3, kc = (q & 7) * 4;
      sA[row][kc + 0] = (__bf16)ra[i].x;  sA[row][kc + 1] = (__bf16)ra[i].y;
      sA[row][kc + 2] = (__bf16)ra[i].z;  sA[row][kc + 3] = (__bf16)ra[i].w;
      int kr = q >> 5, nc = (q & 31) * 4;
      sB[nc + 0][kr] = (__bf16)rb[i].x;  sB[nc + 1][kr] = (__bf16)rb[i].y;
      sB[nc + 2][kr] = (__bf16)rb[i].z;  sB[nc + 3][kr] = (__bf16)rb[i].w;
    }
    __syncthreads();
    // Kick off next tile's global loads; they complete under the WMMAs below.
    if (kt + 1 < Nn / 32) {
      const int k0n = (kt + 1) * 32;
      #pragma unroll
      for (int i = 0; i < 4; ++i) {
        int q = tid + 256 * i;
        ra[i] = *(const float4*)(x + (size_t)(bm + (q >> 3)) * Nn + k0n + (q & 7) * 4);
        rb[i] = *(const float4*)(W + (size_t)(k0n + (q >> 5)) * N3 + bn + (q & 31) * 4);
      }
    }

    // Fragment loads per ISA 16-bit A/B layouts (lane-half selects K range).
    const int off = (lane >> 4) * 8;
    Frag16 a[4], b[2];
    #pragma unroll
    for (int mt = 0; mt < 4; ++mt) {
      int m = wm + mt * 16 + (lane & 15);
      a[mt].q[0] = *(const uint4*)&sA[m][off];
      a[mt].q[1] = *(const uint4*)&sA[m][off + 16];
    }
    #pragma unroll
    for (int nt = 0; nt < 2; ++nt) {
      int n = wn + nt * 16 + (lane & 15);
      b[nt].q[0] = *(const uint4*)&sB[n][(lane >> 4) * 8];
      b[nt].q[1] = *(const uint4*)&sB[n][(lane >> 4) * 8 + 16];
    }
    #pragma unroll
    for (int mt = 0; mt < 4; ++mt)
      #pragma unroll
      for (int nt = 0; nt < 2; ++nt)
        acc[mt][nt] = WMMA_BF16(a[mt].v, b[nt].v, acc[mt][nt]);
  }

  // Epilogue: add bias, scatter bf16 into Q/K/V (B*H, D, 64)
  #pragma unroll
  for (int nt = 0; nt < 2; ++nt) {
    const int colg  = bn + wn + nt * 16 + (lane & 15);
    const float bv  = bias[colg];
    const int three = colg >> 10;          // 0..2 -> Q/K/V
    const int rem   = colg & 1023;
    const int h     = rem >> 6;
    const int pp    = rem & 63;
    __bf16* dst = (three == 0) ? Qb : (three == 1) ? Kb : Vb;
    #pragma unroll
    for (int mt = 0; mt < 4; ++mt) {
      #pragma unroll
      for (int r = 0; r < 8; ++r) {
        int rowg = bm + wm + mt * 16 + r + ((lane >> 4) << 3);
        int bb = rowg >> 11;               // batch
        int dd = rowg & 2047;              // seq position
        dst[((size_t)((bb * Hh + h) * Dd + dd)) * Pp + pp] =
            (__bf16)(acc[mt][nt][r] + bv);
      }
    }
  }
}

// --------------------------------------------------------------------------
// Kernel 2: flash attention per (b,h).  Workgroup = 128 query rows, 8 waves,
// wave = 16 rows x full head.  Online softmax, no NxN materialization.
// Double-buffered K/V tiles; K staged with CDNA5 async global->LDS copies
// (ASYNCcnt) when the builtins are available.
// --------------------------------------------------------------------------
static __device__ __forceinline__ void stage_kv(
    int tid, const __bf16* __restrict__ Kg, const __bf16* __restrict__ Vg,
    __bf16 (*sK)[LDK], __bf16 (*sV)[LDK])
{
  #pragma unroll
  for (int i = 0; i < 2; ++i) {
    int q   = tid + 256 * i;           // 0..511 (8-element chunks)
    int key = q >> 3;
    int pc  = (q & 7) * 8;
#if USE_ASYNC_LDS
    __builtin_amdgcn_global_load_async_to_lds_b128(
        (AS1 v4i_vs*)(Kg + (size_t)key * Pp + pc),
        (AS3 v4i_vs*)&sK[key][pc], 0, 0);
#else
    *(uint4*)&sK[key][pc] = *(const uint4*)(Kg + (size_t)key * Pp + pc);
#endif
    union { uint4 u; __bf16 h[8]; } vv;
    vv.u = *(const uint4*)(Vg + (size_t)key * Pp + pc);
    #pragma unroll
    for (int t = 0; t < 8; ++t) sV[pc + t][key] = vv.h[t];
  }
}

__global__ __launch_bounds__(256) void attn_kernel(
    const __bf16* __restrict__ Qb, const __bf16* __restrict__ Kb,
    const __bf16* __restrict__ Vb, float* __restrict__ out)
{
  __shared__ __align__(16) __bf16 sK2[2][64][LDK];     // [buf][key][p]
  __shared__ __align__(16) __bf16 sV2[2][64][LDK];     // [buf][p][key] (transposed)
  __shared__ __align__(16) __bf16 sP[8][16][LDK];      // per-wave P tile [m][key]

  const int tid  = threadIdx.x;
  const int lane = tid & 31;
  const int wid  = tid >> 5;
  const int bh   = blockIdx.y;                 // b*H + h
  const int d0   = blockIdx.x * 128;
  const size_t base = (size_t)bh * Dd * Pp;
  const __bf16* Kg = Kb + base;
  const __bf16* Vg = Vb + base;

  // Q fragments for this wave's 16 rows stay in registers for the whole kernel
  const int qrow = d0 + wid * 16 + (lane & 15);
  const int off  = (lane >> 4) * 8;
  Frag16 qa[2];
  #pragma unroll
  for (int ks = 0; ks < 2; ++ks) {
    qa[ks].q[0] = *(const uint4*)(Qb + base + (size_t)qrow * Pp + ks * 32 + off);
    qa[ks].q[1] = *(const uint4*)(Qb + base + (size_t)qrow * Pp + ks * 32 + off + 16);
  }

  v8f o[4];
  #pragma unroll
  for (int j = 0; j < 4; ++j) o[j] = v8zero();
  float m_run[8], l_run[8];
  #pragma unroll
  for (int r = 0; r < 8; ++r) { m_run[r] = -1e30f; l_run[r] = 0.f; }

  // Prologue: stage block 0 into buffer 0
  stage_kv(tid, Kg, Vg, sK2[0], sV2[0]);

  for (int kb = 0; kb < Dd / 64; ++kb) {
    const int cur = kb & 1;
#if USE_ASYNC_LDS
    __builtin_amdgcn_s_wait_asynccnt(0);     // my async writes to buf[cur] landed
#endif
    __syncthreads();                         // everyone's staging of buf[cur] visible
    // Issue staging for the next block into the other buffer; completes under
    // the WMMAs/softmax below.
    if (kb + 1 < Dd / 64)
      stage_kv(tid, Kg + (size_t)(kb + 1) * 64 * Pp,
                    Vg + (size_t)(kb + 1) * 64 * Pp, sK2[1 - cur], sV2[1 - cur]);

    __bf16 (*sK)[LDK] = sK2[cur];
    __bf16 (*sV)[LDK] = sV2[cur];

    // S = Q @ K^T : 4 tiles of 16 keys, K=64 contraction in 2 WMMA steps
    v8f s[4];
    #pragma unroll
    for (int j = 0; j < 4; ++j) {
      const int krow = j * 16 + (lane & 15);
      v8f sj = v8zero();
      #pragma unroll
      for (int ks = 0; ks < 2; ++ks) {
        Frag16 bf;
        bf.q[0] = *(const uint4*)&sK[krow][ks * 32 + (lane >> 4) * 8];
        bf.q[1] = *(const uint4*)&sK[krow][ks * 32 + (lane >> 4) * 8 + 16];
        sj = WMMA_BF16(qa[ks].v, bf.v, sj);
      }
      s[j] = sj;
    }

    // Online softmax (rows live in 16-lane halves of C layout)
    #pragma unroll
    for (int r = 0; r < 8; ++r) {
      float mx = fmaxf(fmaxf(s[0][r], s[1][r]), fmaxf(s[2][r], s[3][r]));
      #pragma unroll
      for (int t = 1; t < 16; t <<= 1) mx = fmaxf(mx, __shfl_xor(mx, t, 32));
      const float mnew  = fmaxf(m_run[r], mx);
      const float alpha = __expf(m_run[r] - mnew);
      m_run[r] = mnew;
      float rs = 0.f;
      const int mloc = r + ((lane >> 4) << 3);
      #pragma unroll
      for (int j = 0; j < 4; ++j) {
        float p = __expf(s[j][r] - mnew);
        rs += p;
        sP[wid][mloc][j * 16 + (lane & 15)] = (__bf16)p;
      }
      #pragma unroll
      for (int t = 1; t < 16; t <<= 1) rs += __shfl_xor(rs, t, 32);
      l_run[r] = l_run[r] * alpha + rs;
      #pragma unroll
      for (int j = 0; j < 4; ++j) o[j][r] *= alpha;
    }

    // O += P @ V (wave-private P tile; LDS ops are in-order within a wave)
    #pragma unroll
    for (int ks = 0; ks < 2; ++ks) {
      Frag16 pa;
      const int m = lane & 15;
      pa.q[0] = *(const uint4*)&sP[wid][m][ks * 32 + off];
      pa.q[1] = *(const uint4*)&sP[wid][m][ks * 32 + off + 16];
      #pragma unroll
      for (int j = 0; j < 4; ++j) {
        const int prow = j * 16 + (lane & 15);   // output p column
        Frag16 vf;
        vf.q[0] = *(const uint4*)&sV[prow][ks * 32 + (lane >> 4) * 8];
        vf.q[1] = *(const uint4*)&sV[prow][ks * 32 + (lane >> 4) * 8 + 16];
        o[j] = WMMA_BF16(pa.v, vf.v, o[j]);
      }
    }
  }

  // Finalize and store; reference does a raw (B,H,D,p) -> (B,D,N) reshape,
  // so flat index ((b*H+h)*D + d)*p + pp is exactly the output layout.
  #pragma unroll
  for (int j = 0; j < 4; ++j) {
    const int pp = j * 16 + (lane & 15);
    #pragma unroll
    for (int r = 0; r < 8; ++r) {
      const int d = d0 + wid * 16 + r + ((lane >> 4) << 3);
      out[base + (size_t)d * Pp + pp] = o[j][r] / l_run[r];
    }
  }
}

// --------------------------------------------------------------------------
extern "C" void kernel_launch(void* const* d_in, const int* in_sizes, int n_in,
                              void* d_out, int out_size, void* d_ws, size_t ws_size,
                              hipStream_t stream) {
  (void)in_sizes; (void)n_in; (void)out_size; (void)ws_size;
  const float* x  = (const float*)d_in[0];
  const float* W  = (const float*)d_in[1];
  const float* bq = (const float*)d_in[2];

  const size_t per = (size_t)Bc * Hh * Dd * Pp;   // 4,194,304 elems (8 MB bf16)
  __bf16* Qb = (__bf16*)d_ws;
  __bf16* Kb = Qb + per;
  __bf16* Vb = Kb + per;

  dim3 blk(256);
  dim3 g1(N3 / 128, (Bc * Dd) / 128);   // (24, 32)
  qkv_kernel<<<g1, blk, 0, stream>>>(x, W, bq, Qb, Kb, Vb);

  dim3 g2(Dd / 128, Bc * Hh);           // (16, 32)
  attn_kernel<<<g2, blk, 0, stream>>>(Qb, Kb, Vb, (float*)d_out);
}
